// GATLayer_27187142984375
// MI455X (gfx1250) — compile-verified, matrix operand order checked by reading, not run on hardware
//
#include <hip/hip_runtime.h>

#define F_IN 256
#define HDIM 128
#define LEAKY_SLOPE 0.05f

typedef __attribute__((ext_vector_type(16))) _Float16 v16h;
typedef __attribute__((ext_vector_type(8)))  float    v8f;

// ---------------------------------------------------------------------------
// Kernel 1: h = x @ W (f16 WMMA, f32 accum), fused sl = h@a_w[:128],
//           sr = h@a_w[128:].  One wave per 16-row tile, all 8 col tiles.
// W is converted f32->f16 once per block into LDS, pre-swizzled into the
// 16x16x32 B-fragment layout (lane-contiguous 32B slots -> conflict-free).
// ---------------------------------------------------------------------------
__global__ __launch_bounds__(256) void gat_gemm_kernel(
    const float* __restrict__ x, const float* __restrict__ W,
    const float* __restrict__ a_w, float* __restrict__ h,
    float* __restrict__ sl, float* __restrict__ sr, int N, int nTiles)
{
  // slot s = (kt*8 + ct)*32 + lane, 16 halves per slot = 64 KB total
  __shared__ __align__(32) _Float16 Wlds[8 * 8 * 32 * 16];

  const int tid = threadIdx.x;
  for (int s = tid; s < 2048; s += 256) {
    const int kt = s >> 8;
    const int ct = (s >> 5) & 7;
    const int ln = s & 31;
    const int nl = ln & 15;
    const int g  = ln >> 4;
    const int n  = ct * 16 + nl;
    _Float16* dst = &Wlds[s * 16];
#pragma unroll
    for (int e = 0; e < 16; ++e) {
      // 16-bit B frag element e -> K = (e>>3)*16 + g*8 + (e&7)  (per ISA layout)
      const int k = kt * 32 + ((e >> 3) << 4) + g * 8 + (e & 7);
      dst[e] = (_Float16)W[k * HDIM + n];
    }
  }
  __syncthreads();

  const int wave = tid >> 5;
  const int lane = tid & 31;
  const int tile = blockIdx.x * 8 + wave;
  if (tile >= nTiles) return;

  const int nl = lane & 15;
  const int g  = lane >> 4;
  const int rowBase = tile * 16;

  v8f acc[8] = {};

#pragma unroll
  for (int kt = 0; kt < 8; ++kt) {
    // A fragment: rows rowBase..+15 (m = lane&15), K = kt*32..+31
    v16h a;
    const int arow = min(rowBase + nl, N - 1);
    const float* xrow = x + (size_t)arow * F_IN + kt * 32;
#pragma unroll
    for (int j = 0; j < 8; ++j) {
      // element pair (2j,2j+1): K = (j>>2)*16 + g*8 + (j&3)*2 (+1)
      const int kb = ((j >> 2) << 4) + g * 8 + (j & 3) * 2;
      const float2 f = *(const float2*)(xrow + kb);
      a[2 * j]     = (_Float16)f.x;
      a[2 * j + 1] = (_Float16)f.y;
    }
#pragma unroll
    for (int ct = 0; ct < 8; ++ct) {
      const v16h b = *(const v16h*)&Wlds[((kt * 8 + ct) * 32 + lane) * 16];
      acc[ct] = __builtin_amdgcn_wmma_f32_16x16x32_f16(
          false, a, false, b, (short)0, acc[ct], false, false);
    }
  }

  // per-lane attention coefficients for fused GEMVs
  float awl[8], awr[8];
#pragma unroll
  for (int ct = 0; ct < 8; ++ct) {
    awl[ct] = a_w[ct * 16 + nl];
    awr[ct] = a_w[HDIM + ct * 16 + nl];
  }

  // epilogue: store h tile + reduce sl/sr across the 16-lane half holding each row
#pragma unroll
  for (int r = 0; r < 8; ++r) {
    const int row = rowBase + g * 8 + r;
    float pl = 0.f, pr = 0.f;
#pragma unroll
    for (int ct = 0; ct < 8; ++ct) {
      const float v = acc[ct][r];
      if (row < N) h[(size_t)row * HDIM + ct * 16 + nl] = v;
      pl += v * awl[ct];
      pr += v * awr[ct];
    }
#pragma unroll
    for (int off = 1; off < 16; off <<= 1) {
      pl += __shfl_xor(pl, off, 32);
      pr += __shfl_xor(pr, off, 32);
    }
    if (nl == 0 && row < N) { sl[row] = pl; sr[row] = pr; }
  }
}

// ---------------------------------------------------------------------------
// Kernel 2: per-edge score -> leaky relu -> exp, accumulate per-src denominator
// ---------------------------------------------------------------------------
__global__ __launch_bounds__(256) void edge_score_kernel(
    const int* __restrict__ ei, const float* __restrict__ sl,
    const float* __restrict__ sr, const float* __restrict__ a_b,
    float* __restrict__ evals, float* __restrict__ denom, int E)
{
  const int e = blockIdx.x * blockDim.x + threadIdx.x;
  if (e >= E) return;
  const int s = ei[e];
  const int d = ei[E + e];
  float sc = sl[s] + sr[d] + a_b[0];
  sc = sc > 0.f ? sc : LEAKY_SLOPE * sc;
  const float w = __expf(sc);
  evals[e] = w;
  unsafeAtomicAdd(&denom[s], w);   // global_atomic_add_f32 (L2-side)
}

// ---------------------------------------------------------------------------
// Kernel 3: alpha = e / denom[src]; coeff = edge_vals * alpha (reuses evals)
// ---------------------------------------------------------------------------
__global__ __launch_bounds__(256) void edge_alpha_kernel(
    const int* __restrict__ ei, const float* __restrict__ denom,
    const float* __restrict__ edge_vals, float* __restrict__ evals,
    float* __restrict__ alpha_out, int E)
{
  const int e = blockIdx.x * blockDim.x + threadIdx.x;
  if (e >= E) return;
  const int s = ei[e];
  const float a = evals[e] / denom[s];
  alpha_out[e] = a;
  evals[e] = a * edge_vals[e];
}

// ---------------------------------------------------------------------------
// Kernel 4: out[src] += coeff * h[dst].  Wave per edge: coalesced 512B gather
// of h[dst] (resident in 192MB L2), 4 f32 atomics per lane.
// ---------------------------------------------------------------------------
__global__ __launch_bounds__(256) void aggregate_kernel(
    const int* __restrict__ ei, const float* __restrict__ coeff,
    const float* __restrict__ h, float* __restrict__ out, int E)
{
  const int widx = (blockIdx.x * 256 + threadIdx.x) >> 5;
  const int lane = threadIdx.x & 31;
  if (widx >= E) return;
  const int s = ei[widx];
  const int d = ei[E + widx];
  const float c = coeff[widx];
  const float4 hv = *(const float4*)(h + (size_t)d * HDIM + lane * 4);
  float* o = out + (size_t)s * HDIM + lane * 4;
  unsafeAtomicAdd(o + 0, c * hv.x);
  unsafeAtomicAdd(o + 1, c * hv.y);
  unsafeAtomicAdd(o + 2, c * hv.z);
  unsafeAtomicAdd(o + 3, c * hv.w);
}

// ---------------------------------------------------------------------------
extern "C" void kernel_launch(void* const* d_in, const int* in_sizes, int n_in,
                              void* d_out, int out_size, void* d_ws, size_t ws_size,
                              hipStream_t stream)
{
  (void)n_in; (void)out_size; (void)ws_size;

  const float* x         = (const float*)d_in[0];
  const int*   ei        = (const int*)d_in[1];
  const float* edge_vals = (const float*)d_in[2];
  const float* W         = (const float*)d_in[3];
  const float* a_w       = (const float*)d_in[4];
  const float* a_b       = (const float*)d_in[5];

  const int N = in_sizes[0] / F_IN;
  const int E = in_sizes[2];

  float* out       = (float*)d_out;                 // [N*128]
  float* alpha_out = out + (size_t)N * HDIM;        // [E]

  float* ws    = (float*)d_ws;
  float* h     = ws;                                // N*128
  float* sl    = h + (size_t)N * HDIM;              // N
  float* sr    = sl + N;                            // N
  float* denom = sr + N;                            // N
  float* evals = denom + N;                         // E

  hipMemsetAsync(out,   0, (size_t)N * HDIM * sizeof(float), stream);
  hipMemsetAsync(denom, 0, (size_t)N * sizeof(float), stream);

  const int nTiles = (N + 15) / 16;
  gat_gemm_kernel<<<(nTiles + 7) / 8, 256, 0, stream>>>(x, W, a_w, h, sl, sr, N, nTiles);
  edge_score_kernel<<<(E + 255) / 256, 256, 0, stream>>>(ei, sl, sr, a_b, evals, denom, E);
  edge_alpha_kernel<<<(E + 255) / 256, 256, 0, stream>>>(ei, denom, edge_vals, evals, alpha_out, E);
  aggregate_kernel<<<(E + 7) / 8, 256, 0, stream>>>(ei, evals, h, out, E);
}
